// MoEModel_78615081386104
// MI455X (gfx1250) — matrix operand order repaired and checked by reading, not compiled, not run on hardware
//
#include <hip/hip_runtime.h>
#include <hip/hip_bf16.h>
#include <math.h>

// ---------------- Problem constants (from reference) ----------------
static constexpr int B   = 32768;
static constexpr int D   = 256;
static constexpr int E   = 8;
static constexpr int L1  = 512;
static constexpr int L2  = 256;
static constexpr int L3  = 128;
static constexpr int H   = 64;
static constexpr int T   = 3;
static constexpr float ALPHA  = 0.01f;
static constexpr float LN_EPS = 1e-05f;

static constexpr int TPB = 128;               // 4 waves of 32
static constexpr int MTILE = 16;              // token rows per workgroup tile
static constexpr int MAX_TILES = B / MTILE;

// ---------------- WMMA / vector types ----------------
typedef __attribute__((ext_vector_type(16))) __bf16        v16bf;
typedef __attribute__((ext_vector_type(8)))  float         v8f;
typedef __attribute__((ext_vector_type(4)))  unsigned int  v4u;
typedef __attribute__((ext_vector_type(4)))  unsigned int  u32x4;
typedef __attribute__((ext_vector_type(8)))  int           i32x8;
typedef __attribute__((ext_vector_type(4)))  int           i32x4;

union BfFrag { v16bf v; unsigned short s[16]; v4u q[2]; };

// Tensor Data Mover availability (this toolchain: 6-arg builtin, clang-23 form)
#if defined(__has_builtin)
#if __has_builtin(__builtin_amdgcn_tensor_load_to_lds) && \
    __has_builtin(__builtin_amdgcn_s_wait_tensorcnt)
#define USE_TDM 1
#endif
#endif
#ifndef USE_TDM
#define USE_TDM 0
#endif

// f32 -> bf16 round-to-nearest-even
__device__ __forceinline__ unsigned short f2bf(float f) {
  union { float f; unsigned u; } v; v.f = f;
  unsigned r = v.u + 0x7FFFu + ((v.u >> 16) & 1u);
  return (unsigned short)(r >> 16);
}

#if USE_TDM
// ---------------------------------------------------------------------------
// TDM: 1D contiguous copy (nelem bf16 elements) global -> LDS.
// D# per ISA 8.3/8.4: count=1, gather off; data_size=2B; degenerate 2D tile
// (tile_dim0 = nelem <= 65535, tile_dim1 = 1); type=2 ("image").
// Issued by one wave; completion tracked with TENSORcnt.
// 6-arg builtin: (u32x4 g0, i32x8 g1, i32x4 g2, i32x4 g3, i32x8, i32 cpol)
// ---------------------------------------------------------------------------
__device__ __forceinline__ void tdm_load_1d(unsigned lds_off,
                                            const void* gptr, unsigned nelem)
{
  const unsigned long long ga = (unsigned long long)gptr;
  u32x4 g0 = { 1u,                                   // count=1, no gather
               lds_off,                               // lds_addr
               (unsigned)ga,                          // global_addr[31:0]
               (unsigned)((ga >> 32) & 0x01FFFFFFu)   // global_addr[56:32]
                 | (2u << 30) };                      // type=2
  i32x8 g1 = { (int)0x00010000,                       // data_size=1 (2B)
               (int)((nelem & 0xFFFFu) << 16),        // tensor_dim0[15:0]
               (int)((nelem >> 16) | (1u << 16)),     // dim0 hi | tensor_dim1 lo=1
               (int)(nelem << 16),                    // dim1 hi=0 | tile_dim0
               1,                                     // tile_dim1=1 | tile_dim2=0
               (int)nelem,                            // tensor_dim0_stride[31:0]
               0, 0 };
  i32x4 z4 = { 0, 0, 0, 0 };
  i32x8 z8 = { 0, 0, 0, 0, 0, 0, 0, 0 };
  __builtin_amdgcn_tensor_load_to_lds(g0, g1, z4, z4, z8, 0);
}
#endif

// ---------------------------------------------------------------------------
// Packed-weight layout (bf16, WMMA B-fragment order):
//   dst[ ((kc*ntiles + nt) * 512) + lane*16 + j ]  =  W[k][n]
//   with k = kc*32 + (j<8 ? 0 : 16) + (lane<16 ? 0 : 8) + (j&7) ,  n = nt*16 + (lane&15)
// ---------------------------------------------------------------------------
__global__ void pack_w_kernel(const float* __restrict__ src,
                              unsigned short* __restrict__ dst,
                              int Kd, int Nd, int nmat)
{
  const int id = blockIdx.x * 256 + threadIdx.x;
  const int per = Kd * Nd;
  if (id >= nmat * per) return;
  const int mat = id / per, r = id % per;
  const int ntiles = Nd / 16;
  const int kc  = r / (ntiles * 512);
  const int rem = r % (ntiles * 512);
  const int nt  = rem / 512;
  const int q   = rem % 512;
  const int lane = q >> 4, j = q & 15;
  const int k = kc * 32 + ((j < 8) ? 0 : 16) + ((lane < 16) ? 0 : 8) + (j & 7);
  const int n = nt * 16 + (lane & 15);
  dst[id] = f2bf(src[(size_t)mat * per + (size_t)k * Nd + n]);
}

__global__ void cvt_bf16_kernel(const float* __restrict__ src,
                                unsigned short* __restrict__ dst, size_t n)
{
  const size_t id = (size_t)blockIdx.x * 256 + threadIdx.x;
  if (id < n) dst[id] = f2bf(src[id]);
}

// ---------------------------------------------------------------------------
// Fused 16xN = (16xK) @ (KxN) GEMM, bias + optional LN + ReLU epilogue.
// Weight slabs are TDM-streamed into a double-buffered LDS arena (wslab);
// the f32 epilogue buffer (out) may alias that arena (disjoint lifetimes).
// ---------------------------------------------------------------------------
template <int N, int Kdim, bool DO_LN, bool EMIT_BF16>
__device__ void fused_gemm_tile(const unsigned short* Wp,
                                const float* __restrict__ bias,
                                const float* __restrict__ scale,
                                const float* __restrict__ shift,
                                const unsigned short* Abf,
                                unsigned short* wslab,   // LDS 2*SLAB bf16
                                float* out,              // LDS 16*N f32 (may alias wslab)
                                unsigned short* Aout,
                                float* red)
{
  const int lane = threadIdx.x & 31;
  const int wave = threadIdx.x >> 5;
  constexpr int NT = N / 64;            // 16-wide n-tiles per wave
  constexpr int NTILES = N / 16;
  constexpr int SLAB = NTILES * 512;    // bf16 elems per k-step slab
  constexpr int KSTEPS = Kdim / 32;
  const int col0 = wave * (N / 4);

  v8f acc[NT];
  #pragma unroll
  for (int nt = 0; nt < NT; ++nt)
    #pragma unroll
    for (int r = 0; r < 8; ++r) acc[nt][r] = 0.0f;

  const int m   = lane & 15;
  const int off = (lane < 16) ? 0 : 8;

#if USE_TDM
  const unsigned wslab_off = (unsigned)(unsigned long long)(void*)wslab;
  if (wave == 0)
    tdm_load_1d(wslab_off, Wp, SLAB);

  for (int ks = 0; ks < KSTEPS; ++ks) {
    if (wave == 0) {
      if (ks + 1 < KSTEPS) {          // fill other slab while this one computes
        tdm_load_1d(wslab_off + (unsigned)(((ks + 1) & 1) * SLAB * 2),
                    Wp + (size_t)(ks + 1) * SLAB, SLAB);
        __builtin_amdgcn_s_wait_tensorcnt(1);   // slab ks has landed
      } else {
        __builtin_amdgcn_s_wait_tensorcnt(0);
      }
    }
    __syncthreads();                   // slab ks visible to all waves

    const unsigned short* wsl = wslab + (size_t)(ks & 1) * SLAB;
    const int k0 = ks * 32;

    BfFrag af;
    const v4u* ap = (const v4u*)(Abf + m * Kdim + k0 + off);
    af.q[0] = ap[0];
    af.q[1] = ap[2];

    #pragma unroll
    for (int nt = 0; nt < NT; ++nt) {
      const v4u* bp = (const v4u*)(wsl + ((col0 >> 4) + nt) * 512 + lane * 16);
      BfFrag bf_;
      bf_.q[0] = bp[0];
      bf_.q[1] = bp[1];
      acc[nt] = __builtin_amdgcn_wmma_f32_16x16x32_bf16(
          false, af.v, false, bf_.v, (short)0, acc[nt], false, false);
    }
    __syncthreads();                   // done reading slab ks (reused at ks+2)
  }
#else
  // fallback: B fragments straight from global packed weights
  for (int k0 = 0; k0 < Kdim; k0 += 32) {
    if (k0 + 32 < Kdim)
      __builtin_prefetch((const void*)(Wp +
          ((size_t)((k0 >> 5) + 1) * NTILES + (col0 >> 4)) * 512 + lane * 16), 0, 1);
    BfFrag af;
    const v4u* ap = (const v4u*)(Abf + m * Kdim + k0 + off);
    af.q[0] = ap[0];
    af.q[1] = ap[2];
    #pragma unroll
    for (int nt = 0; nt < NT; ++nt) {
      const v4u* bp = (const v4u*)(Wp +
          ((size_t)(k0 >> 5) * NTILES + (col0 >> 4) + nt) * 512 + lane * 16);
      BfFrag bf_;
      bf_.q[0] = bp[0];
      bf_.q[1] = bp[1];
      acc[nt] = __builtin_amdgcn_wmma_f32_16x16x32_bf16(
          false, af.v, false, bf_.v, (short)0, acc[nt], false, false);
    }
  }
  __syncthreads();                     // before epilogue (out may alias wslab)
#endif

  // ---- epilogue: accumulators (+bias) -> LDS out ----
  #pragma unroll
  for (int nt = 0; nt < NT; ++nt) {
    const int col = col0 + nt * 16 + (lane & 15);
    #pragma unroll
    for (int r = 0; r < 8; ++r) {
      const int row = r + ((lane < 16) ? 0 : 8);
      out[row * N + col] = acc[nt][r] + bias[col];
    }
  }
  __syncthreads();

  if (DO_LN) {
    if (threadIdx.x < 16) {
      const float* rp = out + threadIdx.x * N;
      float mu = 0.0f;
      for (int j = 0; j < N; ++j) mu += rp[j];
      mu /= (float)N;
      float var = 0.0f;
      for (int j = 0; j < N; ++j) { float d = rp[j] - mu; var += d * d; }
      var /= (float)N;
      red[threadIdx.x * 2]     = mu;
      red[threadIdx.x * 2 + 1] = rsqrtf(var + LN_EPS);
    }
    __syncthreads();
    for (int i = threadIdx.x; i < 16 * N; i += TPB) {
      const int row = i / N, col = i % N;
      float h = (out[i] - red[row * 2]) * red[row * 2 + 1] * scale[col] + shift[col];
      h = h > 0.0f ? h : 0.0f;
      if (EMIT_BF16) Aout[i] = f2bf(h);
      else           out[i]  = h;
    }
  } else {
    for (int i = threadIdx.x; i < 16 * N; i += TPB) {
      float h = out[i];
      out[i] = h > 0.0f ? h : 0.0f;
    }
  }
  __syncthreads();
}

// ---------------------------------------------------------------------------
__global__ void init_kernel(int* cnt, float* probsum) {
  if (threadIdx.x < E) { cnt[threadIdx.x] = 0; probsum[threadIdx.x] = 0.0f; }
}

// ---------------------------------------------------------------------------
// gate: one wave per token
// ---------------------------------------------------------------------------
__global__ void gate_kernel(const float* __restrict__ x,
                            const float* __restrict__ gW,
                            const float* __restrict__ gb,
                            int* cnt, float* probsum, int* list, float* topw)
{
  const int tok  = (blockIdx.x * TPB + threadIdx.x) >> 5;
  const int lane = threadIdx.x & 31;
  if (tok >= B) return;

  const float* xr = x + (size_t)tok * D;
  float logit[E];
  #pragma unroll
  for (int e = 0; e < E; ++e) logit[e] = 0.0f;

  #pragma unroll
  for (int i = 0; i < 8; ++i) {
    const int d = lane * 8 + i;
    const float xv = xr[d];
    const float* wr = gW + d * E;
    #pragma unroll
    for (int e = 0; e < E; ++e) logit[e] += xv * wr[e];
  }
  #pragma unroll
  for (int offs = 16; offs > 0; offs >>= 1)
    #pragma unroll
    for (int e = 0; e < E; ++e) logit[e] += __shfl_down(logit[e], offs);

  if (lane == 0) {
    #pragma unroll
    for (int e = 0; e < E; ++e) logit[e] += gb[e];

    float mx = logit[0];
    #pragma unroll
    for (int e = 1; e < E; ++e) mx = fmaxf(mx, logit[e]);
    float p[E], s = 0.0f;
    #pragma unroll
    for (int e = 0; e < E; ++e) { p[e] = __expf(logit[e] - mx); s += p[e]; }
    const float inv = 1.0f / s;
    #pragma unroll
    for (int e = 0; e < E; ++e) atomicAdd(&probsum[e], p[e] * inv);

    int e0 = 0; float l0 = logit[0];
    #pragma unroll
    for (int e = 1; e < E; ++e) if (logit[e] > l0) { l0 = logit[e]; e0 = e; }
    int e1 = (e0 == 0) ? 1 : 0; float l1 = logit[e1];
    #pragma unroll
    for (int e = 0; e < E; ++e)
      if (e != e0 && logit[e] > l1) { l1 = logit[e]; e1 = e; }

    const float w1 = __expf(l1 - l0);
    const float ws = 1.0f + w1;
    topw[2 * tok]     = 1.0f / ws;
    topw[2 * tok + 1] = w1 / ws;

    int p0 = atomicAdd(&cnt[e0], 1); list[e0 * B + p0] = tok * 2;
    int p1 = atomicAdd(&cnt[e1], 1); list[e1 * B + p1] = tok * 2 + 1;
  }
}

__global__ void loss_kernel(const int* cnt, const float* probsum, float* loss_out) {
  if (threadIdx.x == 0) {
    float l = 0.0f;
    for (int e = 0; e < E; ++e)
      l += ((float)cnt[e] / (float)B) * (probsum[e] / (float)B);
    *loss_out = ALPHA * l;
  }
}

// ---------------------------------------------------------------------------
// routed expert MLP. grid = (MAX_TILES, E); early-exit past cnt[e].
// LDS: 64KB weight-stream arena (aliased with f32 epilogue buf) + 16KB bf16 A.
// ---------------------------------------------------------------------------
__global__ void __launch_bounds__(TPB)
expert_kernel(const unsigned short* __restrict__ xbf,
              const unsigned short* __restrict__ W1p, const float* __restrict__ b1,
              const float* __restrict__ s1, const float* __restrict__ t1,
              const unsigned short* __restrict__ W2p, const float* __restrict__ b2,
              const float* __restrict__ s2, const float* __restrict__ t2,
              const unsigned short* __restrict__ W3p, const float* __restrict__ b3,
              const float* __restrict__ s3, const float* __restrict__ t3,
              const int* __restrict__ cnt, const int* __restrict__ list,
              float* __restrict__ sel)
{
  const int e    = blockIdx.y;
  const int tile = blockIdx.x;
  const int count = cnt[e];
  if (tile * MTILE >= count) return;
  const int valid = min(MTILE, count - tile * MTILE);

  // arena: double-buffered weight slab (max 2 * 32*512 bf16 = 64KB),
  // aliased with the 16xN f32 epilogue buffer (lifetimes disjoint).
  __shared__ __align__(16) unsigned char  arena[2 * 32 * 512 * 2];
  __shared__ __align__(16) unsigned short bufA[MTILE * L1];    // 16 KB bf16
  __shared__ float red[32];
  __shared__ int   ents[MTILE];

  unsigned short* wslab = (unsigned short*)arena;
  float*          bufOut = (float*)arena;

  if (threadIdx.x < MTILE) {
    int i = threadIdx.x;
    ents[i] = list[e * B + tile * MTILE + min(i, valid - 1)];
  }
  __syncthreads();

  // gather pre-converted bf16 x rows -> bufA (16B vector copies)
  for (int i = threadIdx.x; i < MTILE * (D / 8); i += TPB) {
    const int row = i / (D / 8), c8 = i % (D / 8);
    const int token = ents[row] >> 1;
    ((v4u*)bufA)[row * (D / 8) + c8] =
        ((const v4u*)(xbf + (size_t)token * D))[c8];
  }
  __syncthreads();

  fused_gemm_tile<L1, D, true, true>(W1p + (size_t)e * D * L1, b1 + e * L1,
                                     s1 + e * L1, t1 + e * L1,
                                     bufA, wslab, bufOut, bufA, red);
  fused_gemm_tile<L2, L1, true, true>(W2p + (size_t)e * L1 * L2, b2 + e * L2,
                                      s2 + e * L2, t2 + e * L2,
                                      bufA, wslab, bufOut, bufA, red);
  fused_gemm_tile<L3, L2, true, false>(W3p + (size_t)e * L2 * L3, b3 + e * L3,
                                       s3 + e * L3, t3 + e * L3,
                                       bufA, wslab, bufOut, nullptr, red);

  for (int i = threadIdx.x; i < MTILE * L3; i += TPB) {
    const int row = i / L3, col = i % L3;
    if (row < valid) sel[(size_t)ents[row] * L3 + col] = bufOut[i];
  }
}

// ---------------------------------------------------------------------------
__global__ void combine_kernel(const float* __restrict__ sel,
                               const float* __restrict__ topw,
                               float* __restrict__ final_)
{
  const size_t id = (size_t)blockIdx.x * blockDim.x + threadIdx.x;
  if (id >= (size_t)B * L3) return;
  const size_t tok = id >> 7;
  const int    col = (int)(id & 127);
  final_[id] = topw[2 * tok]     * sel[(2 * tok)     * L3 + col]
             + topw[2 * tok + 1] * sel[(2 * tok + 1) * L3 + col];
}

// ---------------------------------------------------------------------------
// task heads: WMMA 16x64 K=128 + 64-wide dot
// ---------------------------------------------------------------------------
__global__ void __launch_bounds__(TPB)
head_kernel(const float* __restrict__ final_,
            const unsigned short* __restrict__ hW1p, const float* __restrict__ hb1,
            const float* __restrict__ hW2, const float* __restrict__ hb2,
            float* __restrict__ task_out)
{
  const int t  = blockIdx.y;
  const int b0 = blockIdx.x * MTILE;

  __shared__ __align__(16) unsigned short hslab[2 * 4 * 512];  // 8 KB
  __shared__ __align__(16) unsigned short finb[MTILE * L3];    // 4 KB
  __shared__ __align__(16) float          h[MTILE * H];        // 4 KB

  for (int i = threadIdx.x; i < MTILE * L3; i += TPB)
    finb[i] = f2bf(final_[(size_t)(b0 + i / L3) * L3 + (i % L3)]);
  __syncthreads();

  fused_gemm_tile<H, L3, false, false>(hW1p + (size_t)t * L3 * H, hb1 + t * H,
                                       nullptr, nullptr, finb, hslab, h,
                                       nullptr, nullptr);

  if (threadIdx.x < MTILE) {
    const float* w2 = hW2 + t * H;
    float a = hb2[t];
    #pragma unroll
    for (int j = 0; j < H; ++j) a += h[threadIdx.x * H + j] * w2[j];
    task_out[(size_t)t * B + b0 + threadIdx.x] = a;
  }
}

// ---------------------------------------------------------------------------
extern "C" void kernel_launch(void* const* d_in, const int* in_sizes, int n_in,
                              void* d_out, int out_size, void* d_ws, size_t ws_size,
                              hipStream_t stream)
{
  const float* x      = (const float*)d_in[0];
  const float* gate_W = (const float*)d_in[1];
  const float* gate_b = (const float*)d_in[2];
  const float* W1 = (const float*)d_in[3];
  const float* b1 = (const float*)d_in[4];
  const float* s1 = (const float*)d_in[5];
  const float* t1 = (const float*)d_in[6];
  const float* W2 = (const float*)d_in[7];
  const float* b2 = (const float*)d_in[8];
  const float* s2 = (const float*)d_in[9];
  const float* t2 = (const float*)d_in[10];
  const float* W3 = (const float*)d_in[11];
  const float* b3 = (const float*)d_in[12];
  const float* s3 = (const float*)d_in[13];
  const float* t3 = (const float*)d_in[14];
  const float* hW1 = (const float*)d_in[15];
  const float* hb1 = (const float*)d_in[16];
  const float* hW2 = (const float*)d_in[17];
  const float* hb2 = (const float*)d_in[18];

  // d_out layout: task_out (T*B) | final (B*L3) | lb_loss (1)
  float* task_out = (float*)d_out;
  float* final_   = task_out + (size_t)T * B;
  float* loss_out = final_ + (size_t)B * L3;

  // ---- workspace layout ----
  char* ws = (char*)d_ws;
  size_t o = 0;
  int*   cnt     = (int*)(ws + o);   o += 64;
  float* probsum = (float*)(ws + o); o += 64;
  float* topw    = (float*)(ws + o); o += (size_t)B * 2 * 4;
  int*   list    = (int*)(ws + o);   o += (size_t)E * B * 4;
  float* sel     = (float*)(ws + o); o += (size_t)B * 2 * L3 * 4;
  unsigned short* xbf  = (unsigned short*)(ws + o); o += (size_t)B * D * 2;
  unsigned short* W1p  = (unsigned short*)(ws + o); o += (size_t)E * D  * L1 * 2;
  unsigned short* W2p  = (unsigned short*)(ws + o); o += (size_t)E * L1 * L2 * 2;
  unsigned short* W3p  = (unsigned short*)(ws + o); o += (size_t)E * L2 * L3 * 2;
  unsigned short* hW1p = (unsigned short*)(ws + o); o += (size_t)T * L3 * H  * 2;

  // ---- prologue ----
  init_kernel<<<1, 32, 0, stream>>>(cnt, probsum);
  cvt_bf16_kernel<<<(B * D) / 256, 256, 0, stream>>>(x, xbf, (size_t)B * D);
  pack_w_kernel<<<(E * D  * L1) / 256, 256, 0, stream>>>(W1,  W1p,  D,  L1, E);
  pack_w_kernel<<<(E * L1 * L2) / 256, 256, 0, stream>>>(W2,  W2p,  L1, L2, E);
  pack_w_kernel<<<(E * L2 * L3) / 256, 256, 0, stream>>>(W3,  W3p,  L2, L3, E);
  pack_w_kernel<<<(T * L3 * H ) / 256, 256, 0, stream>>>(hW1, hW1p, L3, H,  T);

  // ---- pipeline ----
  gate_kernel<<<(B * 32) / TPB, TPB, 0, stream>>>(x, gate_W, gate_b,
                                                  cnt, probsum, list, topw);
  loss_kernel<<<1, 32, 0, stream>>>(cnt, probsum, loss_out);
  expert_kernel<<<dim3(MAX_TILES, E), TPB, 0, stream>>>(
      xbf, W1p, b1, s1, t1, W2p, b2, s2, t2, W3p, b3, s3, t3, cnt, list, sel);
  combine_kernel<<<(B * L3) / 256, 256, 0, stream>>>(sel, topw, final_);
  head_kernel<<<dim3(B / MTILE, T), TPB, 0, stream>>>(final_, hW1p, hb1, hW2, hb2,
                                                      task_out);
}
